// ProgressiveBExampleLM_37185826849135
// MI455X (gfx1250) — matrix operand order repaired and checked by reading, not compiled, not run on hardware
//
#include <hip/hip_runtime.h>
#include <hip/hip_bf16.h>
#include <stdint.h>

#define DIM    512
#define SEQ    2048
#define VOCAB  32000
#define BATCH  2
#define WINDOW 4
#define NWIN   (2*WINDOW+1)

static const float S_SC = 0.25f;
static const float B_SC = 0.20f;
static const float C_SC = 0.15f;

// ---------------------------------------------------------------------------
// WMMA GEMM:  C[b] = scale * op(A[b]) x op(B[b]) + bias   (f32 in/out, bf16 MACs)
// Block tile 128x64, 8 waves (4M x 2N), each wave does a 32x32 tile = 4 WMMAs
// per K-step with 2x fragment reuse in registers.
// ---------------------------------------------------------------------------
typedef __attribute__((ext_vector_type(16))) __bf16 v16bf;
typedef __attribute__((ext_vector_type(8)))  float  v8f;

union Frag { v16bf v; unsigned int u[8]; };

#define BM 128
#define BN 64
#define BK 32

__device__ __forceinline__ unsigned short f32_to_bf16_rne(float f) {
  unsigned int u = __float_as_uint(f);
  unsigned int r = u + 0x7FFFu + ((u >> 16) & 1u);
  return (unsigned short)(r >> 16);
}

__global__ void __launch_bounds__(256)
gemm_wmma_kernel(const float* __restrict__ A, const float* __restrict__ B,
                 float* __restrict__ C, const float* __restrict__ bias,
                 int M, int N, int K, int lda, int ldb, int ldc,
                 long strideA, long strideB, long strideC,
                 int transA, int transB, float scale)
{
  __shared__ unsigned short As[BM][BK];   // [m][k], bf16 bits (8 KB)
  __shared__ unsigned short Bs[BN][BK];   // [n][k], bf16 bits (4 KB)

  A += (long)blockIdx.z * strideA;
  B += (long)blockIdx.z * strideB;
  C += (long)blockIdx.z * strideC;

  const int row0 = blockIdx.y * BM;
  const int col0 = blockIdx.x * BN;
  const int tid  = threadIdx.x;
  const int lane = tid & 31;
  const int wid  = tid >> 5;        // 0..7
  const int waveM = wid >> 1;       // 0..3 -> 32 rows each
  const int waveN = wid & 1;        // 0..1 -> 32 cols each
  const int hl  = lane >> 4;        // 0/1
  const int l16 = lane & 15;

  v8f acc[2][2] = {};

  for (int k0 = 0; k0 < K; k0 += BK) {
    // Stage A tile (BM x BK = 4096 elems, 16 per thread), f32 -> bf16
    for (int i = tid; i < BM * BK; i += 256) {
      int m = i >> 5, k = i & 31;
      int gm = row0 + m, gk = k0 + k;
      float v = 0.f;
      if (gm < M && gk < K) {
        v = transA ? A[(long)gk * lda + gm] : A[(long)gm * lda + gk];
        if (gk + BK < K)  // pull next K-tile toward L2/L0 (global_prefetch_b8)
          __builtin_prefetch(transA ? &A[(long)(gk + BK) * lda + gm]
                                    : &A[(long)gm * lda + gk + BK], 0, 3);
      }
      As[m][k] = f32_to_bf16_rne(v);
    }
    // Stage B tile transposed to [n][k] (BN x BK = 2048, 8 per thread)
    for (int i = tid; i < BN * BK; i += 256) {
      int n = i >> 5, k = i & 31;
      int gn = col0 + n, gk = k0 + k;
      float v = 0.f;
      if (gn < N && gk < K)
        v = transB ? B[(long)gn * ldb + gk] : B[(long)gk * ldb + gn];
      Bs[n][k] = f32_to_bf16_rne(v);
    }
    __syncthreads();

    // Fragments per CDNA5 16-bit layouts (packed K-pairs, b128 LDS reads)
    Frag a[2], b[2];
    const int ar0 = waveM * 32 + l16;
    const int bc0 = waveN * 32 + l16;
#pragma unroll
    for (int r = 0; r < 8; ++r) {
      int kkA = ((r >> 2) * 16) + (hl * 8) + ((r & 3) * 2);  // A: 16x32 MxK
      int kkB = ((r >> 2) * 16) + (hl * 8) + ((r & 3) * 2);  // B: 32x16 KxN
      a[0].u[r] = *(const unsigned int*)&As[ar0][kkA];
      a[1].u[r] = *(const unsigned int*)&As[ar0 + 16][kkA];
      b[0].u[r] = *(const unsigned int*)&Bs[bc0][kkB];
      b[1].u[r] = *(const unsigned int*)&Bs[bc0 + 16][kkB];
    }
#pragma unroll
    for (int mt = 0; mt < 2; ++mt)
#pragma unroll
      for (int nt = 0; nt < 2; ++nt)
        acc[mt][nt] = __builtin_amdgcn_wmma_f32_16x16x32_bf16(
            false, a[mt].v, false, b[nt].v, (short)0, acc[mt][nt], false, false);
    __syncthreads();
  }

#pragma unroll
  for (int mt = 0; mt < 2; ++mt) {
#pragma unroll
    for (int nt = 0; nt < 2; ++nt) {
      int ccol = col0 + waveN * 32 + nt * 16 + l16;
      if (ccol < N) {
        float bv = bias ? bias[ccol] : 0.f;
#pragma unroll
        for (int r = 0; r < 8; ++r) {
          int crow = row0 + waveM * 32 + mt * 16 + r + hl * 8;
          if (crow < M)
            C[(long)crow * ldc + ccol] = scale * acc[mt][nt][r] + bv;
        }
      }
    }
  }
}

// ---------------------------------------------------------------------------
// Embedding gather: s = v = embed[tokens]
// ---------------------------------------------------------------------------
__global__ void __launch_bounds__(256)
embed_kernel(const int* __restrict__ tokens, const float* __restrict__ emb,
             float* __restrict__ s, float* __restrict__ v)
{
  int row = blockIdx.x;
  int tid = threadIdx.x;
  long src = (long)tokens[row] * DIM;
  long dst = (long)row * DIM;
  float a = emb[src + tid];
  float b = emb[src + tid + 256];
  s[dst + tid] = a; s[dst + tid + 256] = b;
  v[dst + tid] = a; v[dst + tid + 256] = b;
}

// ---------------------------------------------------------------------------
// Window propagation: one wave32 per (b, n) position
// ---------------------------------------------------------------------------
__global__ void __launch_bounds__(256)
window_prop_kernel(const float* __restrict__ state, const float* __restrict__ val,
                   const float* __restrict__ wp,
                   float* __restrict__ ds, float* __restrict__ dv,
                   int rowsTotal, int N)
{
  int wave = blockIdx.x * 8 + (threadIdx.x >> 5);
  if (wave >= rowsTotal) return;
  int lane = threadIdx.x & 31;
  int b = wave / N;
  int n = wave % N;

  const float* srow = state + (long)wave * DIM;
  float sw[16], ww[16];
#pragma unroll
  for (int i = 0; i < 16; ++i) { sw[i] = srow[lane + 32*i]; ww[i] = wp[lane + 32*i]; }

  float sc[NWIN];
#pragma unroll
  for (int k = 0; k < NWIN; ++k) {
    int idx = n + k - WINDOW;
    bool valid = (idx >= 0) && (idx < N);
    int ic = idx < 0 ? 0 : (idx >= N ? N - 1 : idx);
    const float* nrow = state + ((long)b * N + ic) * DIM;
    float acc = 0.f;
#pragma unroll
    for (int i = 0; i < 16; ++i) acc += sw[i] * ww[i] * nrow[lane + 32*i];
    for (int off = 16; off >= 1; off >>= 1) acc += __shfl_xor(acc, off, 32);
    sc[k] = valid ? acc : -1e9f;
  }
  float mx = sc[0];
#pragma unroll
  for (int k = 1; k < NWIN; ++k) mx = fmaxf(mx, sc[k]);
  float a[NWIN], sum = 0.f;
#pragma unroll
  for (int k = 0; k < NWIN; ++k) { a[k] = __expf(sc[k] - mx); sum += a[k]; }
  float inv = 1.f / sum;

  float os[16] = {}, ov[16] = {};
#pragma unroll
  for (int k = 0; k < NWIN; ++k) {
    int idx = n + k - WINDOW;
    int ic = idx < 0 ? 0 : (idx >= N ? N - 1 : idx);
    const float* nrow = state + ((long)b * N + ic) * DIM;
    const float* vrow = val   + ((long)b * N + ic) * DIM;
    float w = a[k] * inv;
#pragma unroll
    for (int i = 0; i < 16; ++i) {
      os[i] += w * nrow[lane + 32*i];
      ov[i] += w * vrow[lane + 32*i];
    }
  }
  float* dso = ds + (long)wave * DIM;
  float* dvo = dv + (long)wave * DIM;
#pragma unroll
  for (int i = 0; i < 16; ++i) { dso[lane + 32*i] = os[i]; dvo[lane + 32*i] = ov[i]; }
}

// ---------------------------------------------------------------------------
// Stabilizer: s = tanh(s + scale*ds); v = LayerNorm(v + scale*dv) * g + b
// ---------------------------------------------------------------------------
__global__ void __launch_bounds__(256)
stab_kernel(float* __restrict__ s, const float* __restrict__ ds,
            float* __restrict__ v, const float* __restrict__ dv,
            const float* __restrict__ g, const float* __restrict__ bb,
            float scale)
{
  __shared__ float red[8];
  int row = blockIdx.x;
  int tid = threadIdx.x;
  long base = (long)row * DIM;

  float x0 = s[base + tid], x1 = s[base + tid + 256];
  if (ds) { x0 += scale * ds[base + tid]; x1 += scale * ds[base + tid + 256]; }
  s[base + tid]       = tanhf(x0);
  s[base + tid + 256] = tanhf(x1);

  float y0 = v[base + tid], y1 = v[base + tid + 256];
  if (dv) { y0 += scale * dv[base + tid]; y1 += scale * dv[base + tid + 256]; }

  float sum = y0 + y1;
  for (int off = 16; off >= 1; off >>= 1) sum += __shfl_xor(sum, off, 32);
  if ((tid & 31) == 0) red[tid >> 5] = sum;
  __syncthreads();
  float m = 0.f;
#pragma unroll
  for (int i = 0; i < 8; ++i) m += red[i];
  m *= (1.f / (float)DIM);

  float d0 = y0 - m, d1 = y1 - m;
  float vs = d0*d0 + d1*d1;
  for (int off = 16; off >= 1; off >>= 1) vs += __shfl_xor(vs, off, 32);
  __syncthreads();
  if ((tid & 31) == 0) red[tid >> 5] = vs;
  __syncthreads();
  float var = 0.f;
#pragma unroll
  for (int i = 0; i < 8; ++i) var += red[i];
  var *= (1.f / (float)DIM);
  float r = rsqrtf(var + 1e-5f);

  v[base + tid]       = d0 * r * g[tid]       + bb[tid];
  v[base + tid + 256] = d1 * r * g[tid + 256] + bb[tid + 256];
}

// ---------------------------------------------------------------------------
// Top-4 per row (descending, ties -> lower index), one wave32 per row.
// ---------------------------------------------------------------------------
__device__ __forceinline__ void topk_insert(float v[4], int ix[4], float nv, int ni) {
  if ((nv > v[3]) || (nv == v[3] && ni < ix[3])) {
    v[3] = nv; ix[3] = ni;
#pragma unroll
    for (int j = 3; j > 0; --j) {
      bool sw = (v[j] > v[j-1]) || (v[j] == v[j-1] && ix[j] < ix[j-1]);
      if (sw) {
        float tf = v[j]; v[j] = v[j-1]; v[j-1] = tf;
        int   tii = ix[j]; ix[j] = ix[j-1]; ix[j-1] = tii;
      }
    }
  }
}

__global__ void __launch_bounds__(256)
topk4_kernel(const float* __restrict__ logits, float* __restrict__ tv,
             int* __restrict__ ti, int rows, int N)
{
  int wave = blockIdx.x * 8 + (threadIdx.x >> 5);
  if (wave >= rows) return;
  int lane = threadIdx.x & 31;
  const float* row = logits + (long)wave * N;

  float bv[4] = {-3.4e38f, -3.4e38f, -3.4e38f, -3.4e38f};
  int   bi[4] = {0x7fffffff, 0x7fffffff, 0x7fffffff, 0x7fffffff};
  for (int j = lane; j < N; j += 32) topk_insert(bv, bi, row[j], j);

  for (int off = 16; off >= 1; off >>= 1) {
    float ov[4]; int oi[4];
#pragma unroll
    for (int q = 0; q < 4; ++q) {
      ov[q] = __shfl_xor(bv[q], off, 32);
      oi[q] = __shfl_xor(bi[q], off, 32);
    }
#pragma unroll
    for (int q = 0; q < 4; ++q) topk_insert(bv, bi, ov[q], oi[q]);
  }
  if (lane == 0) {
#pragma unroll
    for (int q = 0; q < 4; ++q) {
      tv[(long)wave * 4 + q] = bv[q];
      ti[(long)wave * 4 + q] = bi[q];
    }
  }
}

// ---------------------------------------------------------------------------
// Top-k propagate, part 2: ds/dv[row] = sum_k softmax(tv)_k * {state,val}[ti_k]
// ---------------------------------------------------------------------------
__global__ void __launch_bounds__(256)
topk_gather_kernel(const float* __restrict__ state, const float* __restrict__ val,
                   const float* __restrict__ tv, const int* __restrict__ ti,
                   float* __restrict__ ds, float* __restrict__ dv, int N)
{
  int row = blockIdx.x;                 // [0, B*N)
  int b = row / N;
  int tid = threadIdx.x;
  long tb = (long)row * 4;
  float v0 = tv[tb], v1 = tv[tb+1], v2 = tv[tb+2], v3 = tv[tb+3];
  int   i0 = ti[tb], i1 = ti[tb+1], i2 = ti[tb+2], i3 = ti[tb+3];
  float mx = fmaxf(fmaxf(v0, v1), fmaxf(v2, v3));
  float e0 = __expf(v0-mx), e1 = __expf(v1-mx), e2 = __expf(v2-mx), e3 = __expf(v3-mx);
  float inv = 1.f / (e0 + e1 + e2 + e3);
  float w[4] = {e0*inv, e1*inv, e2*inv, e3*inv};
  int  idx[4] = {i0, i1, i2, i3};

  float a0 = 0, a1 = 0, c0 = 0, c1 = 0;
#pragma unroll
  for (int q = 0; q < 4; ++q) {
    long src = ((long)b * N + idx[q]) * DIM;
    a0 += w[q] * state[src + tid];       a1 += w[q] * state[src + tid + 256];
    c0 += w[q] * val[src + tid];         c1 += w[q] * val[src + tid + 256];
  }
  long o = (long)row * DIM;
  ds[o + tid] = a0; ds[o + tid + 256] = a1;
  dv[o + tid] = c0; dv[o + tid + 256] = c1;
}

// ---------------------------------------------------------------------------
// Sparse transfer, part 2: scatter-add  out[ti_k] += scale*softmax(tv)_k * src
// ---------------------------------------------------------------------------
__global__ void __launch_bounds__(256)
scatter_kernel(const float* __restrict__ src_s, const float* __restrict__ src_v,
               const float* __restrict__ tv, const int* __restrict__ ti,
               float* __restrict__ out_s, float* __restrict__ out_v,
               int S, int ND, float scale)
{
  int row = blockIdx.x;                 // [0, B*S)
  int b = row / S;
  int tid = threadIdx.x;
  long tb = (long)row * 4;
  float v0 = tv[tb], v1 = tv[tb+1], v2 = tv[tb+2], v3 = tv[tb+3];
  int idx[4] = {ti[tb], ti[tb+1], ti[tb+2], ti[tb+3]};
  float mx = fmaxf(fmaxf(v0, v1), fmaxf(v2, v3));
  float e0 = __expf(v0-mx), e1 = __expf(v1-mx), e2 = __expf(v2-mx), e3 = __expf(v3-mx);
  float inv = scale / (e0 + e1 + e2 + e3);
  float w[4] = {e0*inv, e1*inv, e2*inv, e3*inv};

  long sbase = (long)row * DIM;
  float s0 = src_s[sbase + tid], s1 = src_s[sbase + tid + 256];
  float q0 = src_v[sbase + tid], q1 = src_v[sbase + tid + 256];
#pragma unroll
  for (int q = 0; q < 4; ++q) {
    long o = ((long)b * ND + idx[q]) * DIM;
    atomicAdd(&out_s[o + tid],       w[q] * s0);
    atomicAdd(&out_s[o + tid + 256], w[q] * s1);
    atomicAdd(&out_v[o + tid],       w[q] * q0);
    atomicAdd(&out_v[o + tid + 256], w[q] * q1);
  }
}

// ---------------------------------------------------------------------------
// Column-wise softmax over axis=1 (rows S) of logits (B, S, N) -- adapter
// ---------------------------------------------------------------------------
__global__ void __launch_bounds__(256)
softmax_axis1_kernel(float* __restrict__ logits, int S, int N)
{
  __shared__ float red[8];
  int b = blockIdx.y;
  int col = blockIdx.x;
  float* base = logits + (long)b * S * N + col;
  int tid = threadIdx.x;

  float mx = -3.4e38f;
  for (int i = tid; i < S; i += 256) mx = fmaxf(mx, base[(long)i * N]);
  for (int off = 16; off >= 1; off >>= 1) mx = fmaxf(mx, __shfl_xor(mx, off, 32));
  if ((tid & 31) == 0) red[tid >> 5] = mx;
  __syncthreads();
  float gmx = red[0];
#pragma unroll
  for (int i = 1; i < 8; ++i) gmx = fmaxf(gmx, red[i]);

  float sum = 0.f;
  for (int i = tid; i < S; i += 256) sum += __expf(base[(long)i * N] - gmx);
  for (int off = 16; off >= 1; off >>= 1) sum += __shfl_xor(sum, off, 32);
  __syncthreads();
  if ((tid & 31) == 0) red[tid >> 5] = sum;
  __syncthreads();
  float tot = 0.f;
#pragma unroll
  for (int i = 0; i < 8; ++i) tot += red[i];
  float inv = 1.f / tot;
  for (int i = tid; i < S; i += 256)
    base[(long)i * N] = __expf(base[(long)i * N] - gmx) * inv;
}

// ---------------------------------------------------------------------------
// Misc elementwise
// ---------------------------------------------------------------------------
__global__ void __launch_bounds__(256)
fill_kernel(float* __restrict__ p, float v, long n)
{
  long i = (long)blockIdx.x * 256 + threadIdx.x;
  if (i < n) p[i] = v;
}

__global__ void __launch_bounds__(256)
colscale_kernel(float* __restrict__ out, const float* __restrict__ in,
                const float* __restrict__ wp, long n)
{
  long i = (long)blockIdx.x * 256 + threadIdx.x;
  if (i < n) out[i] = in[i] * wp[i & (DIM - 1)];
}

// ---------------------------------------------------------------------------
// Host orchestration
// ---------------------------------------------------------------------------
struct StageCfg { int ne, nc; float alpha, bsb, bbs; };
static const StageCfg CFGS[5] = {
  {2151, 1844, 0.30f, 0.25f, 0.15f},
  {2151, 1844, 0.30f, 0.25f, 0.15f},
  {2253, 1639, 0.65f, 0.55f, 0.35f},
  {2253, 1639, 0.65f, 0.55f, 0.35f},
  {2458, 1434, 1.00f, 0.90f, 0.80f},
};

extern "C" void kernel_launch(void* const* d_in, const int* in_sizes, int n_in,
                              void* d_out, int out_size, void* d_ws, size_t ws_size,
                              hipStream_t stream)
{
  (void)in_sizes; (void)n_in; (void)out_size; (void)ws_size;

  // --- input unpack (dict insertion order flatten) ---
  const int*   tokens  = (const int*)d_in[0];
  const float* p_embed = (const float*)d_in[1];
  const float* p_headw = (const float*)d_in[2];
  const float* p_headb = (const float*)d_in[3];
  auto F = [&](int i) { return (const float*)d_in[i]; };

  // --- workspace layout ---
  const int maxErows = BATCH * 2458;   // 4916
  const int maxCrows = BATCH * 1844;   // 3688
  const int srows    = BATCH * SEQ;    // 4096
  char* wsp = (char*)d_ws;
  auto alloc = [&](size_t bytes) {
    char* p = wsp; wsp += (bytes + 255) & ~(size_t)255; return p;
  };
  float* s      = (float*)alloc((size_t)srows    * DIM * 4);
  float* v      = (float*)alloc((size_t)srows    * DIM * 4);
  float* ds     = (float*)alloc((size_t)maxErows * DIM * 4);
  float* dvb    = (float*)alloc((size_t)maxErows * DIM * 4);
  float* cA_s   = (float*)alloc((size_t)maxCrows * DIM * 4);
  float* cA_v   = (float*)alloc((size_t)maxCrows * DIM * 4);
  float* cB_s   = (float*)alloc((size_t)maxCrows * DIM * 4);
  float* cB_v   = (float*)alloc((size_t)maxCrows * DIM * 4);
  float* e_s    = (float*)alloc((size_t)maxErows * DIM * 4);
  float* e_v    = (float*)alloc((size_t)maxErows * DIM * 4);
  float* scaled = (float*)alloc((size_t)maxErows * DIM * 4);
  float* logits = (float*)alloc((size_t)BATCH * 2458 * 2458 * 4);
  float* tvb    = (float*)alloc((size_t)maxErows * 4 * 4);
  int*   tib    = (int*)  alloc((size_t)maxErows * 4 * 4);

  // --- launch helpers ---
  auto fillz = [&](float* p, long n) {
    fill_kernel<<<(int)((n + 255) / 256), 256, 0, stream>>>(p, 0.f, n);
  };
  auto gemm = [&](const float* A, const float* B, float* C, const float* bias,
                  int M, int N, int K, int lda, int ldb, int ldc,
                  long sA, long sB, long sC, int tA, int tB, float scale, int batch) {
    dim3 grid((N + BN - 1) / BN, (M + BM - 1) / BM, batch);
    gemm_wmma_kernel<<<grid, 256, 0, stream>>>(A, B, C, bias, M, N, K, lda, ldb, ldc,
                                               sA, sB, sC, tA, tB, scale);
  };
  auto stab = [&](float* ps, const float* pds, float* pv, const float* pdv,
                  const float* g, const float* b, float scale, int rows) {
    stab_kernel<<<rows, 256, 0, stream>>>(ps, pds, pv, pdv, g, b, scale);
  };
  auto window = [&](const float* st, const float* vl, const float* wp, int rows) {
    window_prop_kernel<<<(rows + 7) / 8, 256, 0, stream>>>(st, vl, wp, ds, dvb, rows, SEQ);
  };
  auto topk = [&](const float* lg, int rows, int N) {
    topk4_kernel<<<(rows + 7) / 8, 256, 0, stream>>>(lg, tvb, tib, rows, N);
  };
  auto sparse = [&](const float* ss, const float* sv, const float* W, const float* bias,
                    int S, int n, float* out_s, float* out_v, float scale) {
    gemm(ss, W, logits, bias, S, n, DIM, DIM, n, n, (long)S * DIM, 0, (long)S * n,
         0, 0, 1.f, BATCH);
    topk(logits, BATCH * S, n);
    scatter_kernel<<<BATCH * S, 256, 0, stream>>>(ss, sv, tvb, tib, out_s, out_v, S, n, scale);
  };
  auto tprop = [&](const float* st, const float* vl, const float* wp, int N) {
    long tot = (long)BATCH * N * DIM;
    colscale_kernel<<<(int)((tot + 255) / 256), 256, 0, stream>>>(scaled, st, wp, tot);
    gemm(scaled, st, logits, nullptr, N, N, DIM, DIM, DIM, N,
         (long)N * DIM, (long)N * DIM, (long)N * N, 0, 1, 1.f, BATCH);
    topk(logits, BATCH * N, N);
    topk_gather_kernel<<<BATCH * N, 256, 0, stream>>>(st, vl, tvb, tib, ds, dvb, N);
  };

  // --- embed: s = v = embed[tokens] ---
  embed_kernel<<<srows, 256, 0, stream>>>(tokens, p_embed, s, v);

  // --- warmup simple layers ---
  for (int wl = 0; wl < 2; ++wl) {
    const float* wp = F(4 + wl * 3 + 0);
    const float* g  = F(4 + wl * 3 + 1);
    const float* b  = F(4 + wl * 3 + 2);
    window(s, v, wp, srows);
    stab(s, ds, v, dvb, g, b, S_SC, srows);
  }

  // --- full stages ---
  float* c_s = cA_s; float* c_v = cA_v;
  float* n_s = cB_s; float* n_v = cB_v;
  int prev_nc = 0;   // 0 => no compressed state yet

  for (int si = 0; si < 5; ++si) {
    const StageCfg cfg = CFGS[si];
    const int pb = 10 + si * 19;
    const float *w_s_pair = F(pb+0), *w_e_pair = F(pb+1), *w_c_pair = F(pb+2);
    const float *s_g = F(pb+3), *s_b = F(pb+4);
    const float *e_g = F(pb+5), *e_b = F(pb+6);
    const float *c_g = F(pb+7), *c_b = F(pb+8);
    const float *W_expand = F(pb+9),  *b_expand = F(pb+10);
    const float *W_b2s    = F(pb+11), *b_b2s    = F(pb+12);
    const float *W_comp   = F(pb+13), *b_comp   = F(pb+14);
    const float *W_s2b    = F(pb+15), *b_s2b    = F(pb+16);
    const float *W_adapt  = F(pb+17), *b_adapt  = F(pb+18);
    const float sc = cfg.alpha * B_SC;
    const float cb = cfg.alpha * cfg.bbs * C_SC;
    const float cc = cfg.alpha * cfg.bsb * C_SC;
    const int crows = BATCH * cfg.nc;
    const int erows = BATCH * cfg.ne;

    // 1. window prop + stab on (s, v)
    window(s, v, w_s_pair, srows);
    stab(s, ds, v, dvb, s_g, s_b, S_SC, srows);

    // 2. prepare compressed state c
    if (prev_nc == 0) {
      fillz(c_s, (long)crows * DIM);
      fillz(c_v, (long)crows * DIM);
    } else if (prev_nc != cfg.nc) {
      // adapter: r = softmax_S(c @ W_adapt + b); new_c = r^T @ c
      gemm(c_s, W_adapt, logits, b_adapt, prev_nc, cfg.nc, DIM, DIM, cfg.nc, cfg.nc,
           (long)prev_nc * DIM, 0, (long)prev_nc * cfg.nc, 0, 0, 1.f, BATCH);
      softmax_axis1_kernel<<<dim3(cfg.nc, BATCH), 256, 0, stream>>>(logits, prev_nc, cfg.nc);
      gemm(logits, c_s, n_s, nullptr, cfg.nc, DIM, prev_nc, cfg.nc, DIM, DIM,
           (long)prev_nc * cfg.nc, (long)prev_nc * DIM, (long)cfg.nc * DIM, 1, 0, 1.f, BATCH);
      gemm(logits, c_v, n_v, nullptr, cfg.nc, DIM, prev_nc, cfg.nc, DIM, DIM,
           (long)prev_nc * cfg.nc, (long)prev_nc * DIM, (long)cfg.nc * DIM, 1, 0, 1.f, BATCH);
      { float* t; t = c_s; c_s = n_s; n_s = t; t = c_v; c_v = n_v; n_v = t; }
    }
    stab(c_s, nullptr, c_v, nullptr, c_g, c_b, 0.f, crows);

    // 4. expand: e = sc * sparse_trans(c -> ne)
    fillz(e_s, (long)erows * DIM);
    fillz(e_v, (long)erows * DIM);
    sparse(c_s, c_v, W_expand, b_expand, cfg.nc, cfg.ne, e_s, e_v, sc);

    // 5. top-k propagate on e; e = stab(e + sc*dp)
    tprop(e_s, e_v, w_e_pair, cfg.ne);
    stab(e_s, ds, e_v, dvb, e_g, e_b, sc, erows);

    // 6. back-to-seq: s = stab(s + cb * sparse_trans(e -> SEQ))
    fillz(ds,  (long)srows * DIM);
    fillz(dvb, (long)srows * DIM);
    sparse(e_s, e_v, W_b2s, b_b2s, cfg.ne, SEQ, ds, dvb, 1.f);
    stab(s, ds, v, dvb, s_g, s_b, cb, srows);

    // 7+8. n = c + sc * comp(e) + cc * s2b(s)
    hipMemcpyAsync(n_s, c_s, (size_t)crows * DIM * 4, hipMemcpyDeviceToDevice, stream);
    hipMemcpyAsync(n_v, c_v, (size_t)crows * DIM * 4, hipMemcpyDeviceToDevice, stream);
    sparse(e_s, e_v, W_comp, b_comp, cfg.ne, cfg.nc, n_s, n_v, sc);
    sparse(s,   v,   W_s2b,  b_s2b,  SEQ,    cfg.nc, n_s, n_v, cc);

    // 9. top-k propagate on n; n = stab(n + sc*dpc)
    tprop(n_s, n_v, w_c_pair, cfg.nc);
    stab(n_s, ds, n_v, dvb, c_g, c_b, sc, crows);

    // c <- n
    { float* t; t = c_s; c_s = n_s; n_s = t; t = c_v; c_v = n_v; n_v = t; }
    prev_nc = cfg.nc;
  }

  // --- refine simple layers ---
  for (int rl = 0; rl < 2; ++rl) {
    const float* wp = F(105 + rl * 3 + 0);
    const float* g  = F(105 + rl * 3 + 1);
    const float* b  = F(105 + rl * 3 + 2);
    window(s, v, wp, srows);
    stab(s, ds, v, dvb, g, b, S_SC, srows);
  }

  // --- head: out = v[:, SEQ-1, :] @ head_w + head_b  (M=2, lda=SEQ*DIM) ---
  gemm(v + (long)(SEQ - 1) * DIM, p_headw, (float*)d_out, p_headb,
       BATCH, VOCAB, DIM, SEQ * DIM, VOCAB, VOCAB, 0, 0, 0, 0, 0, 1.f, 1);
}
